// SimpleGraphAttentionLayer_79147657331229
// MI455X (gfx1250) — matrix operand order repaired and checked by reading, not compiled, not run on hardware
//
#include <hip/hip_runtime.h>

typedef __attribute__((ext_vector_type(16))) _Float16 v16h;
typedef __attribute__((ext_vector_type(8)))  float    v8f;

#define NN   8192
#define INF_ 512
#define OUTF 128

__device__ __forceinline__ float lrelu(float x) { return x > 0.f ? x : 0.01f * x; }

// ---------------------------------------------------------------------------
// K1: trans_h = h @ W^T + b   (8192x512 @ 512x128), f16 WMMA, f32 accum.
// One wave per 16-row tile; 8 accumulators cover N=0..127.
// ---------------------------------------------------------------------------
__global__ __launch_bounds__(128) void k_trans_h(const float* __restrict__ h,
                                                 const float* __restrict__ W,
                                                 const float* __restrict__ bias,
                                                 float* __restrict__ trans_h) {
  const int lane = threadIdx.x & 31;
  const int wv   = threadIdx.x >> 5;
  const int i0   = (blockIdx.x * 4 + wv) * 16;
  const int M    = lane & 15;
  const int hh   = lane >> 4;

  v8f acc[8] = {};

  for (int kc = 0; kc < INF_; kc += 32) {
    v16h afrag;
#pragma unroll
    for (int r = 0; r < 8; ++r) {
      int k = kc + hh * 16 + 2 * r;
      const float2 hv = *(const float2*)(h + (size_t)(i0 + M) * INF_ + k);
      afrag[2 * r]     = (_Float16)hv.x;
      afrag[2 * r + 1] = (_Float16)hv.y;
    }
#pragma unroll
    for (int t = 0; t < 8; ++t) {
      const int n = t * 16 + (lane & 15);
      v16h bfrag;
#pragma unroll
      for (int r = 0; r < 8; ++r) {
        int k = kc + hh * 16 + 2 * r;
        const float2 wv2 = *(const float2*)(W + (size_t)n * INF_ + k);
        bfrag[2 * r]     = (_Float16)wv2.x;
        bfrag[2 * r + 1] = (_Float16)wv2.y;
      }
      acc[t] = __builtin_amdgcn_wmma_f32_16x16x32_f16(false, afrag, false, bfrag,
                                                      (short)0, acc[t], false, false);
    }
  }
#pragma unroll
  for (int t = 0; t < 8; ++t) {
    const int n  = t * 16 + (lane & 15);
    const float bb = bias[n];
#pragma unroll
    for (int v = 0; v < 8; ++v) {
      const int Mo = v + 8 * hh;
      trans_h[(size_t)(i0 + Mo) * OUTF + n] = acc[t][v] + bb;
    }
  }
}

// ---------------------------------------------------------------------------
// K2: s[i] = dot(trans_h[i,:], a)   (one wave per row)
// ---------------------------------------------------------------------------
__global__ __launch_bounds__(256) void k_s(const float* __restrict__ trans_h,
                                           const float* __restrict__ a,
                                           float* __restrict__ s) {
  const int lane = threadIdx.x & 31;
  const int row  = blockIdx.x * 8 + (threadIdx.x >> 5);
  float p = 0.f;
#pragma unroll
  for (int q = 0; q < 4; ++q) {
    const int n = lane + 32 * q;
    p += trans_h[(size_t)row * OUTF + n] * a[n];
  }
#pragma unroll
  for (int off = 16; off > 0; off >>= 1) p += __shfl_xor(p, off, 32);
  if (lane == 0) s[row] = p;
}

// ---------------------------------------------------------------------------
// K2c: s_max = max_j s[j]  (single block)
// ---------------------------------------------------------------------------
__global__ __launch_bounds__(256) void k_smax(const float* __restrict__ s,
                                              float* __restrict__ smax) {
  __shared__ float red[256];
  float m = -3.0e38f;
  for (int i = threadIdx.x; i < NN; i += 256) m = fmaxf(m, s[i]);
  red[threadIdx.x] = m;
  __syncthreads();
  for (int off = 128; off > 0; off >>= 1) {
    if (threadIdx.x < off) red[threadIdx.x] = fmaxf(red[threadIdx.x], red[threadIdx.x + off]);
    __syncthreads();
  }
  if (threadIdx.x == 0) *smax = red[0];
}

// ---------------------------------------------------------------------------
// K2b: pre-swizzle trans_h into wave32 B-fragment order (f16):
// Bsw[((c*8+t)*32+l)*16 + q] = trans_h[c*32 + (l>>4)*16 + q][t*16 + (l&15)]
// so the fused kernel reads each B fragment as 32 contiguous bytes per lane.
// ---------------------------------------------------------------------------
__global__ __launch_bounds__(256) void k_packB(const float* __restrict__ trans_h,
                                               _Float16* __restrict__ Bsw) {
  const int e = blockIdx.x * 256 + threadIdx.x;      // 0 .. 8192*128-1
  const int q = e & 15;
  const int l = (e >> 4) & 31;
  const int t = (e >> 9) & 7;
  const int c = e >> 12;
  const int k = c * 32 + (l >> 4) * 16 + q;
  const int n = t * 16 + (l & 15);
  Bsw[e] = (_Float16)trans_h[(size_t)k * OUTF + n];
}

// ---------------------------------------------------------------------------
// K3: fused masked-softmax + attention write + WMMA (attention @ trans_h).
// Block = 128 threads (4 waves) per 16-row tile; waves split the 8192 columns.
// ---------------------------------------------------------------------------
__global__ __launch_bounds__(128) void k_gat_fused(const int* __restrict__ adj,
                                                   const float* __restrict__ s,
                                                   const float* __restrict__ smax_p,
                                                   const _Float16* __restrict__ Bsw,
                                                   float* __restrict__ att,
                                                   float* __restrict__ hprime) {
  __shared__ float s_l[NN];           // 32 KB
  __shared__ float num[16 * OUTF];    // 8 KB, cross-wave accumulator
  __shared__ float denomL[4 * 16];
  __shared__ float siL[16], MiL[16], invdL[16];

  const int tid  = threadIdx.x;
  const int lane = tid & 31;
  const int wv   = tid >> 5;
  const int i0   = blockIdx.x * 16;
  const float smax = *smax_p;

  // stage s into LDS; zero the accumulator
  for (int idx = tid; idx < NN / 4; idx += 128)
    ((float4*)s_l)[idx] = ((const float4*)s)[idx];
  for (int e = tid; e < 16 * OUTF; e += 128) num[e] = 0.f;
  if (tid < 16) {
    const float si = s[i0 + tid];
    siL[tid] = si;
    MiL[tid] = lrelu(si + smax);      // upper bound on the row max (lrelu monotone)
  }
  __syncthreads();

  const int cbeg = wv * (NN / 4);
  const int cend = cbeg + (NN / 4);

  // ---- pass 1: row denominators ----
  for (int r = 0; r < 16; ++r) {
    const float sr = siL[r], Mr = MiL[r];
    const int* arow = adj + (size_t)(i0 + r) * NN;
    float sm = 0.f;
    for (int c = cbeg + lane; c < cend; c += 32) {
      __builtin_prefetch(arow + c + 512, 0, 0);
      const int av = arow[c];
      const float x = lrelu(sr + s_l[c]);
      sm += (av > 0) ? __expf(x - Mr) : 0.f;
    }
#pragma unroll
    for (int off = 16; off > 0; off >>= 1) sm += __shfl_xor(sm, off, 32);
    if (lane == 0) denomL[wv * 16 + r] = sm;
  }
  __syncthreads();
  if (tid < 16) {
    const float d = denomL[tid] + denomL[16 + tid] + denomL[32 + tid] + denomL[48 + tid];
    invdL[tid] = (d > 0.f) ? (1.f / d) : 0.f;
  }
  __syncthreads();

  // ---- pass 2: normalized attention + WMMA accumulate ----
  const int M  = lane & 15;
  const int hh = lane >> 4;
  const float sM   = siL[M];
  const float Mm   = MiL[M];
  const float invD = invdL[M];
  const int* arow = adj + (size_t)(i0 + M) * NN;
  float*     orow = att + (size_t)(i0 + M) * NN;

  v8f acc[8] = {};

  for (int cb = cbeg; cb < cend; cb += 32) {
    v16h afrag;
#pragma unroll
    for (int r = 0; r < 8; ++r) {
      const int k = cb + hh * 16 + 2 * r;   // contiguous k order, matches Bsw
      const int2 av = *(const int2*)(arow + k);
      const float x0 = lrelu(sM + s_l[k]);
      const float x1 = lrelu(sM + s_l[k + 1]);
      const float p0 = (av.x > 0) ? __expf(x0 - Mm) * invD : 0.f;
      const float p1 = (av.y > 0) ? __expf(x1 - Mm) * invD : 0.f;
      *(float2*)(orow + k) = make_float2(p0, p1);
      afrag[2 * r]     = (_Float16)p0;
      afrag[2 * r + 1] = (_Float16)p1;
    }
    const v16h* bb = (const v16h*)Bsw + ((size_t)(cb >> 5) * 256) + lane;
#pragma unroll
    for (int t = 0; t < 8; ++t) {
      const v16h bfrag = bb[t * 32];
      acc[t] = __builtin_amdgcn_wmma_f32_16x16x32_f16(false, afrag, false, bfrag,
                                                      (short)0, acc[t], false, false);
    }
  }

  // combine per-wave partials via LDS float atomics (ds_add_f32)
#pragma unroll
  for (int t = 0; t < 8; ++t) {
    const int n = t * 16 + (lane & 15);
#pragma unroll
    for (int v = 0; v < 8; ++v) {
      const int Mo = v + 8 * hh;
      atomicAdd(&num[Mo * OUTF + n], acc[t][v]);
    }
  }
  __syncthreads();

  for (int e = tid; e < 16 * OUTF; e += 128)
    hprime[(size_t)(i0 + (e >> 7)) * OUTF + (e & 127)] = num[e];
}

// ---------------------------------------------------------------------------
extern "C" void kernel_launch(void* const* d_in, const int* in_sizes, int n_in,
                              void* d_out, int out_size, void* d_ws, size_t ws_size,
                              hipStream_t stream) {
  (void)in_sizes; (void)n_in; (void)out_size; (void)ws_size;
  const float* h    = (const float*)d_in[0];
  const float* W    = (const float*)d_in[1];
  const float* bias = (const float*)d_in[2];
  const float* a    = (const float*)d_in[3];
  const int*   adj  = (const int*)d_in[4];

  float* out    = (float*)d_out;
  float* hprime = out;                                  // [8192,128]
  float* att    = out + (size_t)NN * OUTF;              // [8192,8192]

  char* wsb = (char*)d_ws;
  float*    trans_h = (float*)wsb;                                  // 4 MB
  float*    s       = (float*)(wsb + (size_t)NN * OUTF * 4);        // 32 KB
  float*    smax    = s + NN;
  _Float16* Bsw     = (_Float16*)(wsb + (size_t)NN * OUTF * 4 + 65536); // 2 MB

  k_trans_h<<<NN / 16 / 4, 128, 0, stream>>>(h, W, bias, trans_h);
  k_s<<<NN / 8, 256, 0, stream>>>(trans_h, a, s);
  k_smax<<<1, 256, 0, stream>>>(s, smax);
  k_packB<<<(NN * OUTF) / 256, 256, 0, stream>>>(trans_h, Bsw);
  k_gat_fused<<<NN / 16, 128, 0, stream>>>(adj, s, smax, Bsw, att, hprime);
}
